// FusedExpertsNetwork_6571299962932
// MI455X (gfx1250) — matrix operand order repaired and checked by reading, not compiled, not run on hardware
//
#include <hip/hip_runtime.h>
#include <hip/hip_bf16.h>

typedef __attribute__((ext_vector_type(16))) __bf16 v16bf;
typedef __attribute__((ext_vector_type(8)))  __bf16 v8bf;
typedef __attribute__((ext_vector_type(8)))  float  v8f;
typedef int v4i __attribute__((vector_size(16)));
typedef __attribute__((address_space(1))) v4i gv4i;   // global 128-bit chunk
typedef __attribute__((address_space(3))) v4i sv4i;   // LDS 128-bit chunk

#define E_ 8
#define T_ 4096
#define D_ 1024
#define H_ 4096

#define BM 128     // block rows (tokens)
#define BN 256     // block cols
#define BK 32      // K per stage (= one WMMA K)
#define LDT 40     // padded LDS row (bf16 elems): 80B rows keep 16B alignment, break bank strides

#if __has_builtin(__builtin_amdgcn_global_load_async_to_lds_b128)
#define HAVE_ASYNC 1
#else
#define HAVE_ASYNC 0
#endif

#if __has_builtin(__builtin_amdgcn_s_wait_asynccnt)
#define WAIT_ASYNC0() __builtin_amdgcn_s_wait_asynccnt(0)
#else
#define WAIT_ASYNC0() asm volatile("s_wait_asynccnt 0x0" ::: "memory")
#endif

// ---------------------------------------------------------------------------
// f32 -> bf16 conversion prepass (vectorized 4-wide)
// ---------------------------------------------------------------------------
__global__ void __launch_bounds__(256) convert_f32_bf16(const float* __restrict__ in,
                                                        __bf16* __restrict__ out,
                                                        long long n) {
    long long i = ((long long)blockIdx.x * 256 + threadIdx.x) * 4;
    const long long stride = (long long)gridDim.x * 256 * 4;
    for (; i < n; i += stride) {
        float4 f = *(const float4*)(in + i);
        union { __bf16 b[4]; unsigned long long u; } u;
        u.b[0] = (__bf16)f.x; u.b[1] = (__bf16)f.y;
        u.b[2] = (__bf16)f.z; u.b[3] = (__bf16)f.w;
        *(unsigned long long*)(out + i) = u.u;
    }
}

// ---------------------------------------------------------------------------
// Transpose + convert: in [R][C] f32  ->  out [C][R] bf16   (per expert in z)
// Lets GEMM2 use the same K-contiguous staging path as GEMM1.
// ---------------------------------------------------------------------------
__global__ void __launch_bounds__(256) transpose_convert_f32_bf16(
    const float* __restrict__ in, __bf16* __restrict__ out, int R, int C) {
    __shared__ __bf16 tile[32][33];
    const long long base = (long long)blockIdx.z * R * C;
    in  += base;
    out += base;
    const int r0 = blockIdx.y * 32;
    const int c0 = blockIdx.x * 32;
    const int tx = threadIdx.x & 31;
    const int ty = threadIdx.x >> 5;      // 0..7
#pragma unroll
    for (int q = 0; q < 4; ++q) {
        const int r = ty + q * 8;
        tile[r][tx] = (__bf16)in[(long long)(r0 + r) * C + (c0 + tx)];
    }
    __syncthreads();
#pragma unroll
    for (int q = 0; q < 4; ++q) {
        const int r = ty + q * 8;
        out[(long long)(c0 + r) * R + (r0 + tx)] = tile[tx][r];
    }
}

// ---------------------------------------------------------------------------
// Stage a ROWS x BK bf16 tile (K contiguous in memory) into LDS [row][K].
// Async DMA into LDS when available (ASYNCcnt), else b128 load+store fallback.
// ---------------------------------------------------------------------------
template <int ROWS>
__device__ __forceinline__ void stage_tile(const __bf16* __restrict__ g, int ld,
                                           __bf16* l, int k0, int tid) {
    constexpr int CH = (ROWS * BK / 8) / 256;   // b128 chunks per thread
#pragma unroll
    for (int it = 0; it < CH; ++it) {
        const int c  = tid + it * 256;
        const int r  = c >> 2;                  // 4 chunks per 32-elem row
        const int c8 = (c & 3) << 3;
        const __bf16* src = g + (long long)r * ld + (k0 + c8);
        __bf16* dst = l + r * LDT + c8;
#if HAVE_ASYNC
        __builtin_amdgcn_global_load_async_to_lds_b128(
            (gv4i*)src, (sv4i*)dst, 0, 0);
#else
        *(v8bf*)dst = *(const v8bf*)src;
#endif
    }
}

// ---------------------------------------------------------------------------
// C[m,n] = act( sum_k A[m,k] * B[n,k] + bias[n] )      (B stored [N][K])
// Block 128x256, 8 wave32s as 2x4, wave tile 64x64 = 4x4 WMMA fragments.
// Double-buffered LDS, one barrier per K-step, async global->LDS staging.
// ---------------------------------------------------------------------------
template <bool RELU, typename OutT>
__global__ void __launch_bounds__(256) gemm_bf16_wmma(
    const __bf16* __restrict__ Ab, long long sAe, int lda,
    const __bf16* __restrict__ Bb, long long sBe, int ldb,
    const float*  __restrict__ biasb, long long sbe,
    OutT* __restrict__ Ob, long long sOe, int ldo,
    int K)
{
    __shared__ __bf16 sA[2][BM * LDT];   // 2 x 10 KB
    __shared__ __bf16 sB[2][BN * LDT];   // 2 x 20 KB

    const int e = blockIdx.z;
    const __bf16* A    = Ab + (long long)e * sAe + (long long)blockIdx.y * BM * lda;
    const __bf16* B    = Bb + (long long)e * sBe + (long long)blockIdx.x * BN * ldb;
    const float*  bias = biasb + (long long)e * sbe;
    OutT*         O    = Ob + (long long)e * sOe + (long long)blockIdx.y * BM * (long long)ldo;

    const int tid  = threadIdx.x;
    const int lane = tid & 31;
    const int wave = tid >> 5;
    const int wm   = wave & 1;    // 0..1 : 64-row half
    const int wn   = wave >> 1;   // 0..3 : 64-col quarter
    const int lrow = lane & 15;
    const int aklo = (lane < 16) ? 0 : 8;    // A frag: K chunks {aklo..+7, 16+aklo..+7}
    const int bklo = (lane < 16) ? 0 : 8;    // B frag: K chunks {bklo..+7, 16+bklo..+7}

    v8f acc[4][4] = {};

    // prologue: stage k-step 0 into buffer 0
    stage_tile<BM>(A, lda, &sA[0][0], 0, tid);
    stage_tile<BN>(B, ldb, &sB[0][0], 0, tid);

    const int nk = K / BK;
    for (int kt = 0; kt < nk; ++kt) {
        const int cur = kt & 1;
        WAIT_ASYNC0();            // our async stages for 'cur' are done
        __syncthreads();          // everyone's stages done; everyone done reading 'nxt' buffer

        if (kt + 1 < nk) {        // overlap next stage DMA with this step's WMMAs
            const int nxt = cur ^ 1;
            stage_tile<BM>(A, lda, &sA[nxt][0], (kt + 1) * BK, tid);
            stage_tile<BN>(B, ldb, &sB[nxt][0], (kt + 1) * BK, tid);
        }

        v16bf a[4], b[4];
#pragma unroll
        for (int i = 0; i < 4; ++i) {
            const __bf16* p = &sA[cur][(wm * 64 + i * 16 + lrow) * LDT + aklo];
            v8bf lo = *(const v8bf*)(p);
            v8bf hi = *(const v8bf*)(p + 16);
            a[i] = __builtin_shufflevector(lo, hi, 0,1,2,3,4,5,6,7,8,9,10,11,12,13,14,15);
        }
#pragma unroll
        for (int j = 0; j < 4; ++j) {
            const __bf16* p = &sB[cur][(wn * 64 + j * 16 + lrow) * LDT + bklo];
            v8bf lo = *(const v8bf*)(p);
            v8bf hi = *(const v8bf*)(p + 16);
            b[j] = __builtin_shufflevector(lo, hi, 0,1,2,3,4,5,6,7,8,9,10,11,12,13,14,15);
        }

#pragma unroll
        for (int i = 0; i < 4; ++i)
#pragma unroll
            for (int j = 0; j < 4; ++j)
                acc[i][j] = __builtin_amdgcn_wmma_f32_16x16x32_bf16(
                    false, a[i], false, b[j], (short)0, acc[i][j], false, false);
    }

    // epilogue: bias (+ReLU); C/D layout: VGPR r -> M = r + 8*(lane>=16), N = lane&15
    const int n0 = blockIdx.x * BN;
#pragma unroll
    for (int j = 0; j < 4; ++j) {
        const int n = n0 + wn * 64 + j * 16 + (lane & 15);
        const float bv = bias[n];
#pragma unroll
        for (int i = 0; i < 4; ++i) {
#pragma unroll
            for (int r = 0; r < 8; ++r) {
                const int m = wm * 64 + i * 16 + r + ((lane >> 4) << 3);
                float v = acc[i][j][r] + bv;
                if (RELU) v = v > 0.f ? v : 0.f;
                O[(long long)m * ldo + n] = (OutT)v;
            }
        }
    }
}

// ---------------------------------------------------------------------------
extern "C" void kernel_launch(void* const* d_in, const int* in_sizes, int n_in,
                              void* d_out, int out_size, void* d_ws, size_t ws_size,
                              hipStream_t stream) {
    (void)in_sizes; (void)n_in; (void)out_size; (void)ws_size;

    const float* x  = (const float*)d_in[0];
    const float* w1 = (const float*)d_in[1];
    const float* b1 = (const float*)d_in[2];
    const float* w2 = (const float*)d_in[3];
    const float* b2 = (const float*)d_in[4];
    float* out = (float*)d_out;

    const long long nx  = (long long)E_ * T_ * D_;
    const long long nw1 = (long long)E_ * H_ * D_;
    const long long nw2 = nw1;

    __bf16* xb   = (__bf16*)d_ws;        //  64 MB
    __bf16* w1b  = xb + nx;              //  64 MB
    __bf16* w2tb = w1b + nw1;            //  64 MB (transposed: [E][D][H])
    __bf16* yb   = w2tb + nw2;           // 256 MB

    convert_f32_bf16<<<4096, 256, 0, stream>>>(x,  xb,  nx);
    convert_f32_bf16<<<4096, 256, 0, stream>>>(w1, w1b, nw1);
    transpose_convert_f32_bf16<<<dim3(D_ / 32, H_ / 32, E_), 256, 0, stream>>>(w2, w2tb, H_, D_);

    // GEMM1: y = relu(x @ w1^T + b1)   M=T, N=H, K=D ; B = w1 [H][D] = [N][K]
    dim3 g1(H_ / BN, T_ / BM, E_);
    gemm_bf16_wmma<true, __bf16><<<g1, 256, 0, stream>>>(
        xb,  (long long)T_ * D_, D_,
        w1b, (long long)H_ * D_, D_,
        b1,  (long long)H_,
        yb,  (long long)T_ * H_, H_,
        D_);

    // GEMM2: out = y @ w2 + b2         M=T, N=D, K=H ; B = w2t [D][H] = [N][K]
    dim3 g2(D_ / BN, T_ / BM, E_);
    gemm_bf16_wmma<false, float><<<g2, 256, 0, stream>>>(
        yb,   (long long)T_ * H_, H_,
        w2tb, (long long)H_ * D_, H_,
        b2,   (long long)D_,
        out,  (long long)T_ * D_, D_,
        H_);
}
